// DeepseekAttention_7516192768682
// MI455X (gfx1250) — matrix operand order repaired and checked by reading, not compiled, not run on hardware
//
#include <hip/hip_runtime.h>

// ============================================================================
// DeepSeek MLA prefill for gfx1250 (MI455X), bf16 WMMA everywhere.
//
// Shapes: H=128, Q_LORA=1536, KV_LORA=512, ROPE=64, NOPE=128, VD=128,
// HID=5120, QHD=192; b=1, q_len=512, kv_len=2048.
//
// Pipeline (factored; avoids the reference's 1.1GB weight-absorb tensors):
//  1. q_pre = hidden @ Wqa^T                      (512x1536, K=5120)  f32
//  2. q_bf  = rmsnorm(q_pre) * qa_ln_w            -> bf16
//  3. qn[h]   = q_bf @ Wqb_nope[h]^T              (512x128, K=1536)   bf16
//  4. qpe[h]  = rope(q_bf @ Wqb_rope[h]^T)        (512x64,  K=1536)   bf16
//  5. qn2[h]  = qn[h] @ q_absorb[h]               (512x512, K=128)    bf16
//  6. ckv/kpe = bf16(compressed_kv split)
//  7. O[h] = flash_attention(qn2,qpe ; ckv,kpe)   (512x512 latent)    bf16
//  8. u[h] = O[h] @ out_absorb[h]^T               -> U[512,16384]     bf16
//  9. out  = U @ Wo^T                             (512x5120, K=16384) f32
//     BM=512 for stage 9 so Wo (335MB, > L2) streams from HBM exactly once.
// ============================================================================

#define USE_ASYNC_LDS 1

typedef __attribute__((ext_vector_type(16))) __bf16 v16bf;
typedef __attribute__((ext_vector_type(8)))  __bf16 v8bf;
typedef __attribute__((ext_vector_type(4)))  __bf16 v4bf;
typedef __attribute__((ext_vector_type(8)))  float  v8f;

__device__ __forceinline__ __bf16 to_bf16(float x)  { return (__bf16)x; }
__device__ __forceinline__ void store_c(float* p, float v)  { *p = v; }
__device__ __forceinline__ void store_c(__bf16* p, float v) { *p = (__bf16)v; }

__device__ __forceinline__ v8f wmma_bf16(v16bf a, v16bf b, v8f c) {
  return __builtin_amdgcn_wmma_f32_16x16x32_bf16(
      /*neg_a=*/false, a, /*neg_b=*/false, b,
      /*c_mod=*/(short)0, c, /*reuse_a=*/false, /*reuse_b=*/false);
}

#if USE_ASYNC_LDS
// GLOBAL_LOAD_ASYNC_TO_LDS_B128: per-lane 16B global->LDS, tracked by ASYNCcnt.
// LDS byte address = low 32 bits of the generic pointer (aperture truncation).
__device__ __forceinline__ void async_copy16(unsigned lds_addr, const void* gaddr) {
  asm volatile("global_load_async_to_lds_b128 %0, %1, off"
               :: "v"(lds_addr), "v"(gaddr) : "memory");
}
template <int N>
__device__ __forceinline__ void async_waitN() {
  asm volatile("s_wait_asynccnt %0" :: "n"(N) : "memory");
}
__device__ __forceinline__ unsigned lds_addr_of(const void* p) {
  return (unsigned)(uintptr_t)p;
}
#endif

// --- Fragment loaders, exact gfx1250 wave32 layouts (ISA 7.12.2), all via
// --- 16B LDS vector loads (ds_load_b128). -----------------------------------
__device__ __forceinline__ v16bf frag_a(const __bf16* base, int ld, int lane) {
  const int lr = lane & 15, lh = lane >> 4;
  const __bf16* p = base + lr * ld + lh * 8;
  const v8bf lo = *(const v8bf*)(p);
  const v8bf hi = *(const v8bf*)(p + 16);
  return __builtin_shufflevector(lo, hi, 0, 1, 2, 3, 4, 5, 6, 7,
                                 8, 9, 10, 11, 12, 13, 14, 15);
}
__device__ __forceinline__ v16bf frag_b_kmaj(const __bf16* base, int ld, int lane) {
  const int lr = lane & 15, lh = lane >> 4;
  const __bf16* p = base + lr * ld + lh * 8;
  const v8bf lo = *(const v8bf*)(p);
  const v8bf hi = *(const v8bf*)(p + 16);
  return __builtin_shufflevector(lo, hi, 0, 1, 2, 3, 4, 5, 6, 7,
                                 8, 9, 10, 11, 12, 13, 14, 15);
}

// ============================================================================
// Generic tiled WMMA GEMM.  C[M,N] = A[M,K] * op(B).
//   BT=true : B given as [N,K], K contiguous (C = A @ B^T)
//   BT=false: B given as [K,N], N contiguous (C = A @ B)
//   BM in {128, 512}; 256 threads = 8 wave32; tile BM x 64 x 32; wave tile
//   (BM/4) x 32 -> acc[BM/64][2] (BM=512: 128 VGPRs/wave, min B re-streaming).
// bf16 A tiles staged async global->LDS; fp32 operands converted in the fill.
// ============================================================================
template <typename TA, typename TB, typename TC, bool BT, int BM>
__global__ __launch_bounds__(256) void wmma_gemm(
    const TA* __restrict__ Ag, long lda, long strideA,
    const TB* __restrict__ Bg, long ldb, long strideB,
    TC* __restrict__ Cg, long ldc, long strideC,
    int M, int N, int K) {
  constexpr int BN = 64, BK = 32;
  constexpr int WTM = BM / 64;  // 16-row WMMA tiles per wave along M
  __shared__ __bf16 sA[BM * BK];   // [m][k]
  __shared__ __bf16 sB[BN * BK];   // [n][k]  (always stored K-major)

  const int batch = blockIdx.z;
  Ag += (long)batch * strideA;
  Bg += (long)batch * strideB;
  Cg += (long)batch * strideC;

  const int m0 = blockIdx.y * BM, n0 = blockIdx.x * BN;
  const int tid = threadIdx.x, lane = tid & 31, wave = tid >> 5;
  const int lr = lane & 15, lh = lane >> 4;
  const int wm = (wave & 3) * (BM / 4);
  const int wn = (wave >> 2) * 32;

  v8f acc[WTM][2] = {};

  for (long k0 = 0; k0 < K; k0 += BK) {
    // ---- stage A tile (bf16: async or 16B copies; f32: convert) ----
    if constexpr (sizeof(TA) == 2) {
#if USE_ASYNC_LDS
      const unsigned la = lds_addr_of(&sA[0]);
      for (int i = tid; i < BM * BK / 8; i += 256) {
        const int m = i >> 2, g = (i & 3) * 8;
        async_copy16(la + i * 16, &Ag[(long)(m0 + m) * lda + k0 + g]);
      }
#else
      for (int i = tid; i < BM * BK / 8; i += 256)
        ((v8bf*)sA)[i] =
            *(const v8bf*)&Ag[(long)(m0 + (i >> 2)) * lda + k0 + (i & 3) * 8];
#endif
    } else {
      for (int i = tid; i < BM * BK / 4; i += 256) {
        const int m = i >> 3, k4 = (i & 7) * 4;
        const float4 v = *(const float4*)&Ag[(long)(m0 + m) * lda + k0 + k4];
        v4bf o = {to_bf16(v.x), to_bf16(v.y), to_bf16(v.z), to_bf16(v.w)};
        *(v4bf*)&sA[m * BK + k4] = o;
      }
    }
    // ---- stage B tile (fp32 -> bf16 convert) ----
    if constexpr (BT) {
      for (int i = tid; i < BN * BK / 4; i += 256) {
        const int n = i >> 3, k4 = (i & 7) * 4;
        const float4 v = *(const float4*)&Bg[(long)(n0 + n) * ldb + k0 + k4];
        v4bf o = {to_bf16(v.x), to_bf16(v.y), to_bf16(v.z), to_bf16(v.w)};
        *(v4bf*)&sB[n * BK + k4] = o;
      }
    } else {
      for (int i = tid; i < BN * BK / 4; i += 256) {
        const int k = i >> 4, n4 = (i & 15) * 4;  // coalesced along N
        const float4 v = *(const float4*)&Bg[(long)(k0 + k) * ldb + n0 + n4];
        sB[(n4 + 0) * BK + k] = to_bf16(v.x);
        sB[(n4 + 1) * BK + k] = to_bf16(v.y);
        sB[(n4 + 2) * BK + k] = to_bf16(v.z);
        sB[(n4 + 3) * BK + k] = to_bf16(v.w);
      }
    }
    // warm caches for the next K-slab (global_prefetch_b8)
    if (k0 + BK < K) {
      __builtin_prefetch(
          &Ag[(long)(m0 + ((tid * (BM / 128)) & (BM - 1))) * lda + k0 + BK], 0, 0);
      __builtin_prefetch(&Bg[(long)(BT ? (n0 + (tid & 63)) * ldb + k0 + BK
                                       : (k0 + BK + (tid & 31)) * ldb + n0)], 0, 0);
    }
#if USE_ASYNC_LDS
    if constexpr (sizeof(TA) == 2) async_waitN<0>();
#endif
    __syncthreads();

    // ---- WTM x 2 WMMA tiles per wave ----
    const v16bf b0 = frag_b_kmaj(&sB[(wn + 0) * BK], BK, lane);
    const v16bf b1 = frag_b_kmaj(&sB[(wn + 16) * BK], BK, lane);
#pragma unroll
    for (int mi = 0; mi < WTM; ++mi) {
      const v16bf a = frag_a(&sA[(wm + mi * 16) * BK], BK, lane);
      acc[mi][0] = wmma_bf16(a, b0, acc[mi][0]);
      acc[mi][1] = wmma_bf16(a, b1, acc[mi][1]);
    }
    __syncthreads();
  }

  // ---- epilogue: C/D layout = lane holds col lr, VGPR v holds row v+8*lh ----
#pragma unroll
  for (int mi = 0; mi < WTM; ++mi)
#pragma unroll
    for (int ni = 0; ni < 2; ++ni)
#pragma unroll
      for (int v = 0; v < 8; ++v) {
        const int m = m0 + wm + mi * 16 + v + 8 * lh;
        const int n = n0 + wn + ni * 16 + lr;
        store_c(&Cg[(long)m * ldc + n], acc[mi][ni][v]);
      }
}

// ============================================================================
// rmsnorm over 1536 features, one block per row, bf16 out.
// ============================================================================
__global__ __launch_bounds__(256) void rmsnorm_kernel(
    const float* __restrict__ X, const float* __restrict__ w,
    __bf16* __restrict__ Y) {
  const int row = blockIdx.x;
  const float* x = X + (long)row * 1536;
  __shared__ float red[256];
  float s = 0.f;
  for (int i = threadIdx.x; i < 1536; i += 256) { const float v = x[i]; s += v * v; }
  red[threadIdx.x] = s;
  __syncthreads();
  for (int st = 128; st > 0; st >>= 1) {
    if (threadIdx.x < st) red[threadIdx.x] += red[threadIdx.x + st];
    __syncthreads();
  }
  const float inv = rsqrtf(red[0] / 1536.0f + 1e-6f);
  for (int i = threadIdx.x; i < 1536; i += 256)
    Y[(long)row * 1536 + i] = (__bf16)(w[i] * x[i] * inv);
}

// ============================================================================
// RoPE (matches reference apply_rope: interleave reshape -> half-split rotate).
// ============================================================================
__global__ __launch_bounds__(256) void rope_kernel(
    const float* __restrict__ qpe_pre, const int* __restrict__ pos_ids,
    __bf16* __restrict__ qpe_out, int q_len) {
  const long idx = (long)blockIdx.x * 256 + threadIdx.x;
  const long total = (long)128 * q_len * 64;
  if (idx >= total) return;
  const int j = (int)(idx & 63);
  const long hq = idx >> 6;
  const int q = (int)(hq % q_len);
  const float* row = qpe_pre + hq * 64;
  const int p = pos_ids[q];
  const int fi = j & 31;
  const float inv = __powf(10000.0f, -2.0f * (float)fi / 64.0f);
  const float ang = (float)p * inv;
  const float c = __cosf(ang), s = __sinf(ang);
  const float qr  = (j < 32) ? row[2 * j]      : row[2 * (j - 32) + 1];
  const float rot = (j < 32) ? -row[2 * j + 1] : row[2 * (j - 32)];
  qpe_out[idx] = (__bf16)(qr * c + rot * s);
}

// ============================================================================
// Split compressed_kv [kv,576] f32 -> ckv bf16 [kv,512] + kpe bf16 [kv,64].
// ============================================================================
__global__ __launch_bounds__(256) void cvt_kv_kernel(
    const float* __restrict__ in, __bf16* __restrict__ ckv,
    __bf16* __restrict__ kpe, int kv_len) {
  const long idx = (long)blockIdx.x * 256 + threadIdx.x;
  if (idx >= (long)kv_len * 576) return;
  const int col = (int)(idx % 576);
  const long row = idx / 576;
  const __bf16 v = (__bf16)in[idx];
  if (col < 512) ckv[row * 512 + col] = v;
  else           kpe[row * 64 + (col - 512)] = v;
}

// ============================================================================
// Flash attention, one workgroup = (head h, 64 q-rows), 8 wave32.
//  S[64,64] = qn2 @ ckv^T + qpe @ kpe^T  (WMMA, K=576), online softmax (fp32),
//  O[64,512] in VGPRs (wave owns 64-wide latent slice, 4x4 v8f = 128 VGPRs).
//  K/V tiles double-buffered via async global->LDS with thresholded
//  s_wait_asynccnt (18 in-flight b128 ops/wave per tile, in-order retire),
//  overlapping the next tile's HBM fetch with this tile's WMMA work.
//  V transposed once per tile into sKVt[512][64] so every B-fragment is a
//  contiguous ds_load_b128.  Dynamic LDS: 312064 B (< 320KB/WGP on CDNA5).
//  Obuf may alias Qn2 (reads staged to LDS up front; disjoint rows per block).
// ============================================================================
__global__ __launch_bounds__(256) void attn_kernel(
    const __bf16* __restrict__ Qn2, const __bf16* __restrict__ Qpe,
    const __bf16* __restrict__ CKV, const __bf16* __restrict__ KPE,
    __bf16* __restrict__ Obuf, int q_len, int kv_len) {
  extern __shared__ char smem[];
  __bf16* sQn  = (__bf16*)smem;             // 64*512  (65536 B)
  __bf16* sQp  = sQn + 64 * 512;            // 64*64   ( 8192 B)
  __bf16* sKV0 = sQp + 64 * 64;             // 64*512  (65536 B)  [kv][c]
  __bf16* sKV1 = sKV0 + 64 * 512;           // 64*512  (65536 B)
  __bf16* sKp0 = sKV1 + 64 * 512;           // 64*64   ( 8192 B)
  __bf16* sKp1 = sKp0 + 64 * 64;            // 64*64   ( 8192 B)
  __bf16* sKVt = sKp1 + 64 * 64;            // 512*64  (65536 B)  [c][kv]
  float*  sS   = (float*)(sKVt + 512 * 64); // 64*64   (16384 B)
  __bf16* sP   = (__bf16*)(sS + 64 * 64);   // 64*64   ( 8192 B)
  float*  sM   = (float*)(sP + 64 * 64);    // 64
  float*  sL   = sM + 64;                   // 64
  float*  sF   = sL + 64;                   // 64
  __bf16* const sKVb[2] = {sKV0, sKV1};
  __bf16* const sKpb[2] = {sKp0, sKp1};

  const int h = blockIdx.y, q0 = blockIdx.x * 64;
  const int tid = threadIdx.x, lane = tid & 31, wave = tid >> 5;
  const int lr = lane & 15, lh = lane >> 4;
  const int nb = wave * 64;  // this wave's latent-column slice [nb, nb+64)
  const int nkt = kv_len / 64;

  // stage K/V tile kt into buffer `buf`: 16 + 2 = 18 async b128 ops per wave
  auto stage_kv = [&](int kt, int buf) {
    const char* srcKV = (const char*)(CKV + (long)kt * 64 * 512);
    const char* srcKp = (const char*)(KPE + (long)kt * 64 * 64);
#if USE_ASYNC_LDS
    const unsigned lKV = lds_addr_of(sKVb[buf]);
    const unsigned lKp = lds_addr_of(sKpb[buf]);
    for (int i = tid; i < 4096; i += 256) async_copy16(lKV + i * 16, srcKV + i * 16);
    for (int i = tid; i < 512; i += 256)  async_copy16(lKp + i * 16, srcKp + i * 16);
#else
    const ulonglong2* s1 = (const ulonglong2*)srcKV;
    ulonglong2* d1 = (ulonglong2*)sKVb[buf];
    for (int i = tid; i < 4096; i += 256) d1[i] = s1[i];
    const ulonglong2* s2 = (const ulonglong2*)srcKp;
    ulonglong2* d2 = (ulonglong2*)sKpb[buf];
    for (int i = tid; i < 512; i += 256) d2[i] = s2[i];
#endif
  };

  // ---- stage this block's Q tiles (contiguous rows) ----
  {
    const char* srcQn = (const char*)(Qn2 + ((long)h * q_len + q0) * 512);
    const char* srcQp = (const char*)(Qpe + ((long)h * q_len + q0) * 64);
#if USE_ASYNC_LDS
    const unsigned lQn = lds_addr_of(sQn), lQp = lds_addr_of(sQp);
    for (int i = tid; i < 4096; i += 256) async_copy16(lQn + i * 16, srcQn + i * 16);
    for (int i = tid; i < 512; i += 256)  async_copy16(lQp + i * 16, srcQp + i * 16);
#else
    const ulonglong2* s1 = (const ulonglong2*)srcQn; ulonglong2* d1 = (ulonglong2*)sQn;
    for (int i = tid; i < 4096; i += 256) d1[i] = s1[i];
    const ulonglong2* s2 = (const ulonglong2*)srcQp; ulonglong2* d2 = (ulonglong2*)sQp;
    for (int i = tid; i < 512; i += 256) d2[i] = s2[i];
#endif
  }
  stage_kv(0, 0);  // prologue: tile 0 in flight alongside Q
  if (tid < 64) { sM[tid] = -3.0e38f; sL[tid] = 0.f; }

  v8f acc[4][4] = {};
  const float sc = 0.07216878364870323f;  // rsqrt(192)

  for (int kt = 0; kt < nkt; ++kt) {
    const int cur = kt & 1;
#if USE_ASYNC_LDS
    if (kt + 1 < nkt) {
      stage_kv(kt + 1, cur ^ 1);  // issue next tile, then wait only for current
      async_waitN<18>();
    } else {
      async_waitN<0>();
    }
#endif
    __syncthreads();  // all waves' current-tile data visible
    const __bf16* sKV = sKVb[cur];
    const __bf16* sKp = sKpb[cur];

    // ---- build transposed V tile sKVt[c][kv] from sKV[kv][c] ----
    for (int i = tid; i < 4096; i += 256) {
      const int kv = i >> 6, c8 = (i & 63) * 8;
      const v8bf v = ((const v8bf*)sKV)[i];
#pragma unroll
      for (int e = 0; e < 8; ++e) sKVt[(c8 + e) * 64 + kv] = v[e];
    }
    __syncthreads();

    // ---- S = Qn2 @ CKV^T + Qpe @ KPE^T  (16 16x16 tiles, 2 per wave) ----
#pragma unroll
    for (int s = 0; s < 2; ++s) {
      const int t = wave * 2 + s, tm = t >> 2, tn = t & 3;
      v8f sacc = {};
      for (int kk = 0; kk < 16; ++kk) {  // 512 latent dims
        const v16bf a = frag_a(&sQn[(tm * 16) * 512 + kk * 32], 512, lane);
        const v16bf b = frag_b_kmaj(&sKV[(tn * 16) * 512 + kk * 32], 512, lane);
        sacc = wmma_bf16(a, b, sacc);
      }
#pragma unroll
      for (int kk = 0; kk < 2; ++kk) {   // 64 rope dims
        const v16bf a = frag_a(&sQp[(tm * 16) * 64 + kk * 32], 64, lane);
        const v16bf b = frag_b_kmaj(&sKp[(tn * 16) * 64 + kk * 32], 64, lane);
        sacc = wmma_bf16(a, b, sacc);
      }
#pragma unroll
      for (int v = 0; v < 8; ++v)
        sS[(tm * 16 + v + 8 * lh) * 64 + tn * 16 + lr] = sacc[v];
    }
    __syncthreads();

    // ---- online softmax over this kv tile (one thread per q-row) ----
    if (tid < 64) {
      const float mOld = sM[tid];
      float mNew = mOld;
      for (int j = 0; j < 64; ++j) mNew = fmaxf(mNew, sS[tid * 64 + j] * sc);
      const float f = __expf(mOld - mNew);
      float sum = 0.f;
      for (int j = 0; j < 64; ++j) {
        const float p = __expf(sS[tid * 64 + j] * sc - mNew);
        sP[tid * 64 + j] = (__bf16)p;
        sum += p;
      }
      sL[tid] = sL[tid] * f + sum;
      sM[tid] = mNew;
      sF[tid] = f;
    }
    __syncthreads();

    // ---- rescale running O, then O += P @ V (B-fragments from sKVt) ----
#pragma unroll
    for (int mt = 0; mt < 4; ++mt)
#pragma unroll
      for (int v = 0; v < 8; ++v) {
        const float f = sF[mt * 16 + v + 8 * lh];
#pragma unroll
        for (int nt = 0; nt < 4; ++nt) acc[mt][nt][v] *= f;
      }
#pragma unroll
    for (int kk = 0; kk < 2; ++kk) {  // 64 kv positions
      v16bf pa[4];
#pragma unroll
      for (int mt = 0; mt < 4; ++mt)
        pa[mt] = frag_a(&sP[(mt * 16) * 64 + kk * 32], 64, lane);
#pragma unroll
      for (int nt = 0; nt < 4; ++nt) {
        const v16bf bv = frag_b_kmaj(&sKVt[(nb + nt * 16) * 64 + kk * 32], 64, lane);
#pragma unroll
        for (int mt = 0; mt < 4; ++mt)
          acc[mt][nt] = wmma_bf16(pa[mt], bv, acc[mt][nt]);
      }
    }
    __syncthreads();  // end of iteration: sKVt/sS/sP reusable, buffers swappable
  }

  // ---- normalize and write O (bf16) ----
#pragma unroll
  for (int mt = 0; mt < 4; ++mt)
#pragma unroll
    for (int v = 0; v < 8; ++v) {
      const int m = mt * 16 + v + 8 * lh;
      const float inv = 1.0f / sL[m];
#pragma unroll
      for (int nt = 0; nt < 4; ++nt) {
        const int n = nb + nt * 16 + lr;
        Obuf[((long)h * q_len + q0 + m) * 512 + n] = (__bf16)(acc[mt][nt][v] * inv);
      }
    }
}

// ============================================================================
// Host launcher
// ============================================================================
extern "C" void kernel_launch(void* const* d_in, const int* in_sizes, int n_in,
                              void* d_out, int out_size, void* d_ws, size_t ws_size,
                              hipStream_t stream) {
  const float* hidden = (const float*)d_in[0];   // [q_len, 5120]
  const int*   pos    = (const int*)d_in[1];     // [1, q_len]
  const float* ckv_in = (const float*)d_in[2];   // [kv_len, 576]
  const float* Wqa    = (const float*)d_in[3];   // [1536, 5120]
  const float* ln_w   = (const float*)d_in[4];   // [1536]
  const float* Wqb    = (const float*)d_in[5];   // [128*192, 1536]
  const float* Wkvb   = (const float*)d_in[6];   // [128*256, 512]
  const float* Wo     = (const float*)d_in[7];   // [5120, 128*128]
  float* out = (float*)d_out;                    // [q_len, 5120]

  const int q_len  = in_sizes[0] / 5120;         // 512
  const int kv_len = in_sizes[2] / 576;          // 2048

  char* w = (char*)d_ws;
  auto alloc = [&](size_t bytes) -> char* {
    char* p = w;
    w += (bytes + 255) & ~(size_t)255;
    return p;
  };
  float*  q_pre   = (float*) alloc((size_t)q_len * 1536 * 4);
  __bf16* q_bf    = (__bf16*)alloc((size_t)q_len * 1536 * 2);
  float*  qpe_pre = (float*) alloc((size_t)128 * q_len * 64 * 4);
  __bf16* qpe     = (__bf16*)alloc((size_t)128 * q_len * 64 * 2);
  __bf16* qn      = (__bf16*)alloc((size_t)128 * q_len * 128 * 2);
  __bf16* qn2     = (__bf16*)alloc((size_t)128 * q_len * 512 * 2);
  __bf16* ckv_bf  = (__bf16*)alloc((size_t)kv_len * 512 * 2);
  __bf16* kpe_bf  = (__bf16*)alloc((size_t)kv_len * 64 * 2);
  __bf16* U       = (__bf16*)qpe_pre;  // alias: dead after rope; sizes match
  __bf16* Obuf    = qn2;               // alias: disjoint rows per block

  // 1) q_pre = hidden @ Wqa^T  (BM=128 for block-count; Wqa is L2-resident)
  wmma_gemm<float, float, float, true, 128>
      <<<dim3(1536 / 64, q_len / 128, 1), 256, 0, stream>>>(
      hidden, 5120, 0, Wqa, 5120, 0, q_pre, 1536, 0, q_len, 1536, 5120);

  // 2) rmsnorm -> bf16
  rmsnorm_kernel<<<q_len, 256, 0, stream>>>(q_pre, ln_w, q_bf);

  // 3) qn[h] = q_bf @ Wqb_nope[h]^T
  wmma_gemm<__bf16, float, __bf16, true, 512>
      <<<dim3(2, q_len / 512, 128), 256, 0, stream>>>(
      q_bf, 1536, 0, Wqb, 1536, 192L * 1536, qn, 128, (long)q_len * 128,
      q_len, 128, 1536);

  // 4) qpe_pre[h] = q_bf @ Wqb_rope[h]^T
  wmma_gemm<__bf16, float, float, true, 512>
      <<<dim3(1, q_len / 512, 128), 256, 0, stream>>>(
      q_bf, 1536, 0, Wqb + 128L * 1536, 1536, 192L * 1536, qpe_pre, 64,
      (long)q_len * 64, q_len, 64, 1536);

  // 5) rope -> bf16
  {
    const long total = (long)128 * q_len * 64;
    rope_kernel<<<(unsigned)((total + 255) / 256), 256, 0, stream>>>(
        qpe_pre, pos, qpe, q_len);
  }
  // 6) split/convert kv cache
  {
    const long total = (long)kv_len * 576;
    cvt_kv_kernel<<<(unsigned)((total + 255) / 256), 256, 0, stream>>>(
        ckv_in, ckv_bf, kpe_bf, kv_len);
  }

  // 7) qn2[h] = qn[h] @ q_absorb[h]   (B is [K=128, N=512], N-major)
  wmma_gemm<__bf16, float, __bf16, false, 512>
      <<<dim3(8, q_len / 512, 128), 256, 0, stream>>>(
      qn, 128, (long)q_len * 128, Wkvb, 512, 256L * 512, qn2, 512,
      (long)q_len * 512, q_len, 512, 128);

  // 8) flash attention
  {
    const int ATT_SMEM = 312064;
    hipFuncSetAttribute(reinterpret_cast<const void*>(attn_kernel),
                        hipFuncAttributeMaxDynamicSharedMemorySize, ATT_SMEM);
    attn_kernel<<<dim3(q_len / 64, 128), 256, ATT_SMEM, stream>>>(
        qn2, qpe, ckv_bf, kpe_bf, Obuf, q_len, kv_len);
  }

  // 9) u[h] = O[h] @ out_absorb[h]^T -> U[q, h*128+v]
  wmma_gemm<__bf16, float, __bf16, true, 512>
      <<<dim3(2, q_len / 512, 128), 256, 0, stream>>>(
      Obuf, 512, (long)q_len * 512, Wkvb + 128L * 512, 512, 256L * 512,
      U, 16384, 128, q_len, 128, 512);

  // 10) out = U @ Wo^T  (BM=512: Wo streams from HBM exactly once)
  wmma_gemm<__bf16, float, float, true, 512>
      <<<dim3(5120 / 64, q_len / 512, 1), 256, 0, stream>>>(
      U, 16384, 0, Wo, 16384, 0, out, 5120, 0, q_len, 5120, 16384);

  (void)n_in; (void)out_size; (void)ws_size;
}
